// HierarchicalGroupStageMoEv3_41841571398187
// MI455X (gfx1250) — matrix operand order repaired and checked by reading, not compiled, not run on hardware
//
#include <hip/hip_runtime.h>
#include <hip/hip_bf16.h>

// ---- problem constants (from reference) ----
#define Nn   4096   // B*L tokens
#define Dd   512
#define Gg   8
#define Ss   4
#define FGf  4
#define FEe  64
#define Ee   32
#define DHh  256
#define RHh  256
#define SHARPc 16.0f

typedef __attribute__((ext_vector_type(16))) __bf16 v16bf;
typedef __attribute__((ext_vector_type(8)))  float  v8f;
typedef __attribute__((ext_vector_type(4)))  unsigned int v4u;

union Frag { v16bf bf; v4u u[2]; };

__device__ __forceinline__ unsigned short f2bf(float f) {
    unsigned int u = __float_as_uint(f);
    u += 0x7fffu + ((u >> 16) & 1u);          // round-to-nearest-even
    return (unsigned short)(u >> 16);
}
__device__ __forceinline__ float gelu_f(float x) {
    return 0.5f * x * (1.0f + erff(x * 0.70710678118654752440f));
}

// ------------------------------------------------------------------
// K0a: straight f32 -> bf16 convert
__global__ __launch_bounds__(256) void cvt_bf16_kernel(const float* __restrict__ in,
                                                       unsigned short* __restrict__ out, int n) {
    int i = blockIdx.x * 256 + threadIdx.x;
    if (i < n) out[i] = f2bf(in[i]);
}

// K0b: batched transpose+convert: in [b][R][C] f32 -> out [b][C][R] bf16
__global__ __launch_bounds__(256) void tconv_kernel(const float* __restrict__ in,
                                                    unsigned short* __restrict__ out,
                                                    int R, int C, long total) {
    long i = (long)blockIdx.x * 256 + threadIdx.x;
    if (i >= total) return;
    long rc = (long)R * C;
    long b = i / rc, rem = i - b * rc;
    long c = rem / R, r = rem - c * R;
    out[i] = f2bf(in[b * rc + r * C + c]);
}

// ------------------------------------------------------------------
// K1: gh = gelu(hidden @ Wr1 + br1)   [4096x512] x [512x256] via WMMA bf16
// one block = one 16-token tile; 8 waves x 2 n-tiles (16 cols each) = 256 cols
__global__ __launch_bounds__(256) void gemm_gh_kernel(const unsigned short* __restrict__ hidbf,
                                                      const unsigned short* __restrict__ wr1t, // [RH][D]
                                                      const float* __restrict__ br1,
                                                      float* __restrict__ gh) {
    __shared__ __align__(16) unsigned short X[16 * Dd];
    const int tile = blockIdx.x, tid = threadIdx.x;
    for (int c = tid; c < 16 * Dd / 8; c += 256) {           // 16B chunks
        int r = c >> 6, k = (c & 63) * 8;
        *(v4u*)&X[r * Dd + k] = *(const v4u*)&hidbf[((size_t)(tile * 16 + r)) * Dd + k];
    }
    __syncthreads();
    const int wave = tid >> 5, lane = tid & 31;
    const int rowA = lane & 15, khalf = (lane >> 4) * 16;    // A frag: contig 16 K (matches B)
    const int nB = lane & 15, kB = (lane >> 4) * 16;          // B frag: lane=col, contig 16 K
    v8f acc0 = {0,0,0,0,0,0,0,0}, acc1 = {0,0,0,0,0,0,0,0};
    const int n0 = wave * 32 + nB, n1 = n0 + 16;
    for (int k0 = 0; k0 < Dd; k0 += 32) {
        Frag a, b0, b1;
        a.u[0]  = *(const v4u*)&X[rowA * Dd + k0 + khalf];
        a.u[1]  = *(const v4u*)&X[rowA * Dd + k0 + 8 + khalf];
        b0.u[0] = *(const v4u*)&wr1t[(size_t)n0 * Dd + k0 + kB];
        b0.u[1] = *(const v4u*)&wr1t[(size_t)n0 * Dd + k0 + kB + 8];
        b1.u[0] = *(const v4u*)&wr1t[(size_t)n1 * Dd + k0 + kB];
        b1.u[1] = *(const v4u*)&wr1t[(size_t)n1 * Dd + k0 + kB + 8];
        acc0 = __builtin_amdgcn_wmma_f32_16x16x32_bf16(false, a.bf, false, b0.bf, (short)0, acc0, false, false);
        acc1 = __builtin_amdgcn_wmma_f32_16x16x32_bf16(false, a.bf, false, b1.bf, (short)0, acc1, false, false);
    }
    const int col0 = wave * 32 + (lane & 15), col1 = col0 + 16;
    const float bb0 = br1[col0], bb1 = br1[col1];
    const int rbase = (lane >> 4) * 8;
#pragma unroll
    for (int r = 0; r < 8; ++r) {
        int row = tile * 16 + rbase + r;
        gh[(size_t)row * RHh + col0] = gelu_f(acc0[r] + bb0);
        gh[(size_t)row * RHh + col1] = gelu_f(acc1[r] + bb1);
    }
}

// ------------------------------------------------------------------
// K2: router. one wave per token; lane = (g,s) pair for ilog.
__global__ __launch_bounds__(256) void router_kernel(const float* __restrict__ hidden,
                                                     const float* __restrict__ feat,
                                                     const float* __restrict__ gh,
                                                     const float* __restrict__ Wr2,
                                                     const float* __restrict__ br2,
                                                     const float* __restrict__ Wfe,
                                                     const float* __restrict__ bfe,
                                                     const float* __restrict__ Win_h,
                                                     const float* __restrict__ Win_f,
                                                     const float* __restrict__ bin_b,
                                                     int* __restrict__ tok_e,
                                                     float* __restrict__ tok_w) {
    __shared__ float sHid[8][Dd];
    __shared__ float sGh[8][RHh];
    __shared__ float sFeat[8][Gg * FGf];
    __shared__ float sGlog[8][Gg];
    __shared__ float sIlog[8][Gg * Ss];
    __shared__ int   sSel[8][2];
    __shared__ float sGw[8][2];
    const int tid = threadIdx.x, wave = tid >> 5, lane = tid & 31;
    const int t = blockIdx.x * 8 + wave;
    for (int i = lane; i < Dd; i += 32)  sHid[wave][i] = hidden[(size_t)t * Dd + i];
    for (int i = lane; i < RHh; i += 32) sGh[wave][i]  = gh[(size_t)t * RHh + i];
    sFeat[wave][lane] = feat[(size_t)t * (Gg * FGf) + lane];
    __syncthreads();
    if (lane < Gg) {
        float a = br2[lane];
        for (int r = 0; r < RHh; ++r) a += sGh[wave][r] * Wr2[r * Gg + lane];
        sGlog[wave][lane] = a;
    }
    __syncthreads();
    if (lane == 0) {                                   // top-2 groups + softmax
        int i1 = 0; float v1 = sGlog[wave][0];
        for (int g = 1; g < Gg; ++g) if (sGlog[wave][g] > v1) { v1 = sGlog[wave][g]; i1 = g; }
        int i2 = -1; float v2 = -3.4e38f;
        for (int g = 0; g < Gg; ++g) if (g != i1 && sGlog[wave][g] > v2) { v2 = sGlog[wave][g]; i2 = g; }
        float e2 = __expf(v2 - v1), inv = 1.0f / (1.0f + e2);
        sSel[wave][0] = i1; sSel[wave][1] = i2;
        sGw[wave][0] = inv; sGw[wave][1] = e2 * inv;
    }
    __syncthreads();
    {   // ilog per (g,s)
        const int g = lane >> 2, s = lane & 3;
        float acc = bin_b[g * Ss + s];
        for (int d = 0; d < Dd; ++d)
            acc += sHid[wave][d] * Win_h[((size_t)g * Dd + d) * Ss + s];
        for (int e = 0; e < FEe; ++e) {
            float fe = bfe[g * FEe + e];
#pragma unroll
            for (int f = 0; f < FGf; ++f)
                fe += sFeat[wave][g * FGf + f] * Wfe[((size_t)(g * FGf + f)) * FEe + e];
            acc += fe * Win_f[((size_t)g * FEe + e) * Ss + s];
        }
        float score = 0.0f;
#pragma unroll
        for (int f = 0; f < FGf; ++f) {
            float x = sFeat[wave][g * FGf + f];
            score += fminf(fmaxf(x, 0.0f), 1.0f);      // uniform[0,1) -> clamp branch of _to_ratio
        }
        score *= 0.25f;
        float center = (float)s * (1.0f / 3.0f);
        float diff = score - center;
        acc += -SHARPc * diff * diff;                  // BIAS=1, TEMP=1
        sIlog[wave][lane] = acc;
    }
    __syncthreads();
    if (lane < 2) {                                    // top-2 stages within each selected group
        int g = sSel[wave][lane];
        float gw = sGw[wave][lane];
        float v[4];
#pragma unroll
        for (int s = 0; s < 4; ++s) v[s] = sIlog[wave][g * 4 + s];
        int j1 = 0; float v1 = v[0];
#pragma unroll
        for (int s = 1; s < 4; ++s) if (v[s] > v1) { v1 = v[s]; j1 = s; }
        int j2 = -1; float v2 = -3.4e38f;
#pragma unroll
        for (int s = 0; s < 4; ++s) if (s != j1 && v[s] > v2) { v2 = v[s]; j2 = s; }
        float e2 = __expf(v2 - v1), inv = 1.0f / (1.0f + e2);
        int jb = lane * 2;
        tok_e[t * 4 + jb]     = g * 4 + j1;  tok_w[t * 4 + jb]     = gw * inv;
        tok_e[t * 4 + jb + 1] = g * 4 + j2;  tok_w[t * 4 + jb + 1] = gw * e2 * inv;
    }
}

// ------------------------------------------------------------------
// K3: deterministic per-expert compaction (block prefix scan in token order)
__global__ __launch_bounds__(256) void compact_kernel(const int* __restrict__ tok_e,
                                                      int* __restrict__ elist,
                                                      int* __restrict__ tok_slot,
                                                      int* __restrict__ cnt) {
    const int e = blockIdx.x, tid = threadIdx.x;
    __shared__ int sc[256];
    __shared__ int sBase;
    if (tid == 0) sBase = 0;
    __syncthreads();
    for (int iter = 0; iter < Nn / 256; ++iter) {
        int t = iter * 256 + tid;
        int match = -1;
#pragma unroll
        for (int j = 0; j < 4; ++j) if (tok_e[t * 4 + j] == e) match = j;
        int m = (match >= 0) ? 1 : 0;
        sc[tid] = m; __syncthreads();
        for (int ofs = 1; ofs < 256; ofs <<= 1) {
            int v = (tid >= ofs) ? sc[tid - ofs] : 0;
            __syncthreads();
            sc[tid] += v;
            __syncthreads();
        }
        int pos = sBase + sc[tid] - m;
        if (m) { elist[e * Nn + pos] = t; tok_slot[t * 4 + match] = pos; }
        __syncthreads();
        if (tid == 0) sBase += sc[255];
        __syncthreads();
    }
    if (tid == 0) cnt[e] = sBase;
}

__global__ void scan_kernel(const int* __restrict__ cnt, int* __restrict__ basep) {
    if (threadIdx.x == 0) {
        int a = 0;
        for (int e = 0; e < Ee; ++e) { basep[e] = a; a += cnt[e]; }
    }
}

// ------------------------------------------------------------------
// K4: fused expert double-GEMM. block = (expert, 16-token tile of its list).
// GEMM1: h1 = gelu(X[16x512] @ We1[e] + be1)   (WMMA bf16, 8 waves x 2 n-tiles)
// GEMM2: Y  = h1[16x256] @ We2[e]              (WMMA bf16, 8 waves x 4 n-tiles)
// Y rows stored unweighted into staging (row = base[e] + local idx).
__global__ __launch_bounds__(256) void gemm_expert_kernel(const unsigned short* __restrict__ hidbf,
                                                          const unsigned short* __restrict__ we1t, // [E][DH][D]
                                                          const unsigned short* __restrict__ we2t, // [E][D][DH]
                                                          const float* __restrict__ be1,
                                                          const int* __restrict__ elist,
                                                          const int* __restrict__ cnt,
                                                          const int* __restrict__ basep,
                                                          float* __restrict__ stage) {
    const int e = blockIdx.x >> 8;
    const int tile = blockIdx.x & 255;
    const int n_rows = cnt[e];
    const int row0 = tile * 16;
    if (row0 >= n_rows) return;
    __shared__ __align__(16) unsigned short X[16 * Dd];
    __shared__ __align__(16) unsigned short H1[16 * DHh];
    const int tid = threadIdx.x;
    for (int c = tid; c < 16 * Dd / 8; c += 256) {         // gather 16 rows as 16B chunks
        int r = c >> 6, k = (c & 63) * 8;
        int rr = row0 + r;
        int tok = elist[e * Nn + (rr < n_rows ? rr : (n_rows - 1))];
        *(v4u*)&X[r * Dd + k] = *(const v4u*)&hidbf[(size_t)tok * Dd + k];
    }
    __syncthreads();
    const int wave = tid >> 5, lane = tid & 31;
    const int rowA = lane & 15, khalf = (lane >> 4) * 16;
    const int nB = lane & 15, kB = (lane >> 4) * 16;
    // ---- GEMM1 ----
    v8f acc0 = {0,0,0,0,0,0,0,0}, acc1 = {0,0,0,0,0,0,0,0};
    const unsigned short* B1 = &we1t[(size_t)e * DHh * Dd];
    const int n0 = wave * 32 + nB, n1 = n0 + 16;
    for (int k0 = 0; k0 < Dd; k0 += 32) {
        Frag a, b0, b1;
        a.u[0]  = *(const v4u*)&X[rowA * Dd + k0 + khalf];
        a.u[1]  = *(const v4u*)&X[rowA * Dd + k0 + 8 + khalf];
        b0.u[0] = *(const v4u*)&B1[(size_t)n0 * Dd + k0 + kB];
        b0.u[1] = *(const v4u*)&B1[(size_t)n0 * Dd + k0 + kB + 8];
        b1.u[0] = *(const v4u*)&B1[(size_t)n1 * Dd + k0 + kB];
        b1.u[1] = *(const v4u*)&B1[(size_t)n1 * Dd + k0 + kB + 8];
        acc0 = __builtin_amdgcn_wmma_f32_16x16x32_bf16(false, a.bf, false, b0.bf, (short)0, acc0, false, false);
        acc1 = __builtin_amdgcn_wmma_f32_16x16x32_bf16(false, a.bf, false, b1.bf, (short)0, acc1, false, false);
    }
    {   // epilogue: +be1, gelu, bf16 -> LDS H1
        const int col0 = wave * 32 + (lane & 15), col1 = col0 + 16;
        const float bb0 = be1[e * DHh + col0], bb1 = be1[e * DHh + col1];
        const int rbase = (lane >> 4) * 8;
#pragma unroll
        for (int r = 0; r < 8; ++r) {
            int row = rbase + r;
            H1[row * DHh + col0] = f2bf(gelu_f(acc0[r] + bb0));
            H1[row * DHh + col1] = f2bf(gelu_f(acc1[r] + bb1));
        }
    }
    __syncthreads();
    // ---- GEMM2 ----
    v8f c[4];
#pragma unroll
    for (int j = 0; j < 4; ++j) c[j] = (v8f){0,0,0,0,0,0,0,0};
    const unsigned short* B2 = &we2t[(size_t)e * Dd * DHh];
    for (int k0 = 0; k0 < DHh; k0 += 32) {
        Frag a;
        a.u[0] = *(const v4u*)&H1[rowA * DHh + k0 + khalf];
        a.u[1] = *(const v4u*)&H1[rowA * DHh + k0 + 8 + khalf];
#pragma unroll
        for (int j = 0; j < 4; ++j) {
            int m = (wave * 4 + j) * 16 + nB;
            Frag b;
            b.u[0] = *(const v4u*)&B2[(size_t)m * DHh + k0 + kB];
            b.u[1] = *(const v4u*)&B2[(size_t)m * DHh + k0 + kB + 8];
            c[j] = __builtin_amdgcn_wmma_f32_16x16x32_bf16(false, a.bf, false, b.bf, (short)0, c[j], false, false);
        }
    }
    {   // scatter to staging (unweighted; weight applied in reduce)
        const int baseE = basep[e];
        const int rbase = (lane >> 4) * 8;
#pragma unroll
        for (int j = 0; j < 4; ++j) {
            int col = (wave * 4 + j) * 16 + (lane & 15);
            v8f cc = c[j];
#pragma unroll
            for (int r = 0; r < 8; ++r) {
                int idx = row0 + rbase + r;
                if (idx < n_rows)
                    stage[(size_t)(baseE + idx) * Dd + col] = cc[r];
            }
        }
    }
}

// ------------------------------------------------------------------
// K5: y[t] = sum_j w_j * (stage[slot_j] + be2[e_j])
__global__ __launch_bounds__(256) void reduce_kernel(const float* __restrict__ stage,
                                                     const float* __restrict__ be2,
                                                     const int* __restrict__ tok_e,
                                                     const float* __restrict__ tok_w,
                                                     const int* __restrict__ tok_slot,
                                                     const int* __restrict__ basep,
                                                     float* __restrict__ y) {
    const int t = blockIdx.x, tid = threadIdx.x;
    __shared__ int se[4], ss[4];
    __shared__ float sw[4];
    if (tid < 4) {
        int e = tok_e[t * 4 + tid];
        se[tid] = e;
        sw[tid] = tok_w[t * 4 + tid];
        ss[tid] = basep[e] + tok_slot[t * 4 + tid];
    }
    __syncthreads();
    for (int d = tid; d < Dd; d += 256) {
        float a = 0.0f;
#pragma unroll
        for (int j = 0; j < 4; ++j)
            a += sw[j] * (stage[(size_t)ss[j] * Dd + d] + be2[(size_t)se[j] * Dd + d]);
        y[(size_t)t * Dd + d] = a;
    }
}

// ------------------------------------------------------------------
extern "C" void kernel_launch(void* const* d_in, const int* in_sizes, int n_in,
                              void* d_out, int out_size, void* d_ws, size_t ws_size,
                              hipStream_t stream) {
    (void)in_sizes; (void)n_in; (void)out_size; (void)ws_size;
    const float* hidden = (const float*)d_in[0];
    const float* feat   = (const float*)d_in[1];
    // d_in[2] valid_mask: unused (deleted in reference)
    const float* Wr1   = (const float*)d_in[3];
    const float* br1   = (const float*)d_in[4];
    const float* Wr2   = (const float*)d_in[5];
    const float* br2   = (const float*)d_in[6];
    const float* Wfe   = (const float*)d_in[7];
    const float* bfe   = (const float*)d_in[8];
    const float* Win_h = (const float*)d_in[9];
    const float* Win_f = (const float*)d_in[10];
    const float* bin_b = (const float*)d_in[11];
    const float* We1   = (const float*)d_in[12];
    const float* be1   = (const float*)d_in[13];
    const float* We2   = (const float*)d_in[14];
    const float* be2   = (const float*)d_in[15];

    char* w = (char*)d_ws;
    auto nxt = [&](size_t bytes) { char* p = w; w += (bytes + 255) & ~(size_t)255; return p; };
    unsigned short* hid_bf   = (unsigned short*)nxt((size_t)Nn * Dd * 2);
    unsigned short* wr1t     = (unsigned short*)nxt((size_t)RHh * Dd * 2);
    unsigned short* we1t     = (unsigned short*)nxt((size_t)Ee * DHh * Dd * 2);
    unsigned short* we2t     = (unsigned short*)nxt((size_t)Ee * Dd * DHh * 2);
    float*          gh       = (float*)nxt((size_t)Nn * RHh * 4);
    int*            tok_e    = (int*)nxt((size_t)Nn * 4 * 4);
    float*          tok_w    = (float*)nxt((size_t)Nn * 4 * 4);
    int*            tok_slot = (int*)nxt((size_t)Nn * 4 * 4);
    int*            cntp     = (int*)nxt((size_t)Ee * 4);
    int*            basep    = (int*)nxt((size_t)Ee * 4);
    int*            elist    = (int*)nxt((size_t)Ee * Nn * 4);
    float*          stage    = (float*)nxt((size_t)4 * Nn * Dd * 4);

    cvt_bf16_kernel<<<(Nn * Dd) / 256, 256, 0, stream>>>(hidden, hid_bf, Nn * Dd);
    tconv_kernel<<<(Dd * RHh) / 256, 256, 0, stream>>>(Wr1, wr1t, Dd, RHh, (long)Dd * RHh);
    tconv_kernel<<<(Ee * Dd * DHh) / 256, 256, 0, stream>>>(We1, we1t, Dd, DHh, (long)Ee * Dd * DHh);
    tconv_kernel<<<(Ee * DHh * Dd) / 256, 256, 0, stream>>>(We2, we2t, DHh, Dd, (long)Ee * DHh * Dd);

    gemm_gh_kernel<<<Nn / 16, 256, 0, stream>>>(hid_bf, wr1t, br1, gh);
    router_kernel<<<Nn / 8, 256, 0, stream>>>(hidden, feat, gh, Wr2, br2, Wfe, bfe,
                                              Win_h, Win_f, bin_b, tok_e, tok_w);
    compact_kernel<<<Ee, 256, 0, stream>>>(tok_e, elist, tok_slot, cntp);
    scan_kernel<<<1, 32, 0, stream>>>(cntp, basep);
    gemm_expert_kernel<<<Ee * (Nn / 16), 256, 0, stream>>>(hid_bf, we1t, we2t, be1,
                                                           elist, cntp, basep, stage);
    reduce_kernel<<<Nn, 256, 0, stream>>>(stage, be2, tok_e, tok_w, tok_slot, basep,
                                          (float*)d_out);
}
